// LocationSensitiveAttention_42271068127883
// MI455X (gfx1250) — matrix operand order repaired and checked by reading, not compiled, not run on hardware
//
#include <hip/hip_runtime.h>
#include <hip/hip_bf16.h>
#include <math.h>

// ---------------------------------------------------------------------------
// LocationSensitiveAttention, fused for MI455X (gfx1250, wave32, WMMA).
//
// Roofline: two passes over `memory` (2 x 262 MB @ 23.3 TB/s ~= 22.5 us) is
// the floor; the 16.8 GFLOP pm-GEMM is done with v_wmma_f32_16x16x32_bf16
// (f32 accumulate) so compute stays under the HBM bound. The 31-tap conv and
// the Wl projection are algebraically collapsed into one K=62 bf16 GEMM.
// ---------------------------------------------------------------------------

typedef __attribute__((ext_vector_type(16))) __bf16 v16bf;
typedef __attribute__((ext_vector_type(8)))  float  v8f;

#define B_   64
#define T_   2000
#define QD   1024
#define MD   512
#define AD   128
#define NF   32
#define KS   31
#define NTT  8              // AD / 16  (N tiles)
#define KCM  16             // MD / 32  (K chunks, memory GEMM)
#define KCL  2              // ceil(62/32) (K chunks, location GEMM)
#define TILES_PER_B 125     // T / 16
#define TOTAL_TILES (B_ * TILES_PER_B)
#define WPB  4              // waves per block in score kernel
#define TCH  8              // t-chunks for context partials
#define TPC  (T_ / TCH)

// ---------------- prep: pq[b][a] = query@Wq + bq + bm (biases folded) ------
__global__ __launch_bounds__(AD)
void lsa_pq_kernel(const float* __restrict__ query, const float* __restrict__ Wq,
                   const float* __restrict__ bq, const float* __restrict__ bm,
                   float* __restrict__ pq) {
  const int b = blockIdx.x, a = threadIdx.x;
  float acc = bq[a] + bm[a];
  const float* qr = query + (size_t)b * QD;
  for (int k = 0; k < QD; ++k) acc = fmaf(qr[k], Wq[k * AD + a], acc);
  pq[b * AD + a] = acc;
}

// ---------------- prep: Wm -> bf16 B-fragments [nt][kc][lane][e] -----------
// B-matrix (32x16 bf16) layout: lane L holds column n=L%16; its 16 K values
// are K = (e&7) + (e>=8?16:0) + 8*(L/16)  (matches the A fragment K order).
__global__ __launch_bounds__(256)
void lsa_wmfrag_kernel(const float* __restrict__ Wm, unsigned short* __restrict__ frag_) {
  __bf16* frag = (__bf16*)frag_;
  const int idx  = blockIdx.x * 256 + threadIdx.x;      // 65536 total
  const int e    = idx & 15;
  const int lane = (idx >> 4) & 31;
  const int kc   = (idx >> 9) & 15;
  const int nt   = idx >> 13;
  const int d    = kc * 32 + (e & 7) + (e >= 8 ? 16 : 0) + 8 * (lane >> 4);
  const int a    = nt * 16 + (lane & 15);
  frag[idx] = (__bf16)Wm[d * AD + a];
}

// ---------------- prep: W2[j][a] = sum_f conv_w[f][c][k]*Wl[f][a] ----------
// j = c*31 + k (0..61), zero-padded to K=64; fragmented like Wm.
__global__ __launch_bounds__(256)
void lsa_w2frag_kernel(const float* __restrict__ conv_w, const float* __restrict__ Wl,
                       unsigned short* __restrict__ frag_) {
  __bf16* frag = (__bf16*)frag_;
  const int idx  = blockIdx.x * 256 + threadIdx.x;      // 8192 total
  const int e    = idx & 15;
  const int lane = (idx >> 4) & 31;
  const int kc   = (idx >> 9) & 1;
  const int nt   = idx >> 10;
  const int j    = kc * 32 + (e & 7) + (e >= 8 ? 16 : 0) + 8 * (lane >> 4);
  const int a    = nt * 16 + (lane & 15);
  float v = 0.f;
  if (j < 2 * KS) {
    const int c = j / KS, k = j - c * KS;
    for (int f = 0; f < NF; ++f)
      v = fmaf(conv_w[(f * 2 + c) * KS + k], Wl[f * AD + a], v);
  }
  frag[idx] = (__bf16)v;
}

// ---------------- fused score kernel: one wave per 16-row t-tile -----------
__global__ __launch_bounds__(32 * WPB)
void lsa_score_kernel(const float* __restrict__ memory,
                      const float* __restrict__ aligncat,
                      const float* __restrict__ pq,
                      const unsigned short* __restrict__ wmfrag_,
                      const unsigned short* __restrict__ w2frag_,
                      const float* __restrict__ v_w,
                      const float* __restrict__ v_b,
                      float* __restrict__ scores) {
  const int lane = threadIdx.x & 31;
  const int wave = threadIdx.x >> 5;
  const int tile = blockIdx.x * WPB + wave;             // exact: 8000 tiles
  const int b    = tile / TILES_PER_B;
  const int t0   = (tile - b * TILES_PER_B) * 16;
  const int h    = lane >> 4;                            // lane half
  const int n    = lane & 15;                            // A row m / B,C col n

  v8f acc[NTT];
  #pragma unroll
  for (int i = 0; i < NTT; ++i) acc[i] = (v8f){0.f,0.f,0.f,0.f,0.f,0.f,0.f,0.f};

  // ---- pm tile: memory[b, t0..t0+15, :] @ Wm  (bf16 WMMA, f32 acc) ----
  // A-matrix 16x32 bf16 layout: lane holds row m=n; K(e,h) = e+(e>=8?8:0)+8h
  // -> two contiguous 8-float (32B) row chunks per fragment.
  const v16bf* wm = (const v16bf*)wmfrag_;
  const float* arow = memory + (size_t)(b * T_ + t0 + n) * MD;
  for (int kc = 0; kc < KCM; ++kc) {
    const int d0 = kc * 32 + 8 * h;
    const float4 f0 = *(const float4*)(arow + d0);
    const float4 f1 = *(const float4*)(arow + d0 + 4);
    const float4 f2 = *(const float4*)(arow + d0 + 16);
    const float4 f3 = *(const float4*)(arow + d0 + 20);
    v16bf a;
    a[0]=(__bf16)f0.x;  a[1]=(__bf16)f0.y;  a[2]=(__bf16)f0.z;  a[3]=(__bf16)f0.w;
    a[4]=(__bf16)f1.x;  a[5]=(__bf16)f1.y;  a[6]=(__bf16)f1.z;  a[7]=(__bf16)f1.w;
    a[8]=(__bf16)f2.x;  a[9]=(__bf16)f2.y;  a[10]=(__bf16)f2.z; a[11]=(__bf16)f2.w;
    a[12]=(__bf16)f3.x; a[13]=(__bf16)f3.y; a[14]=(__bf16)f3.z; a[15]=(__bf16)f3.w;
    #pragma unroll
    for (int nt = 0; nt < NTT; ++nt) {
      const v16bf bb = wm[(nt * KCM + kc) * 32 + lane];
      acc[nt] = __builtin_amdgcn_wmma_f32_16x16x32_bf16(
          false, a, false, bb, (short)0, acc[nt], false, false);
    }
  }

  // ---- pl tile: im2col(alignments_cat) @ W2  (K=62 padded to 64) ----
  const v16bf* w2 = (const v16bf*)w2frag_;
  #pragma unroll
  for (int kc = 0; kc < KCL; ++kc) {
    v16bf a;
    #pragma unroll
    for (int e = 0; e < 16; ++e) {
      const int j = kc * 32 + e + (e >= 8 ? 8 : 0) + 8 * h;
      float v = 0.f;
      if (j < 2 * KS) {
        const int c  = j / KS, k = j - c * KS;
        const int ts = t0 + n + k - (KS - 1) / 2;
        if (ts >= 0 && ts < T_) v = aligncat[(size_t)(b * T_ + ts) * 2 + c];
      }
      a[e] = (__bf16)v;
    }
    #pragma unroll
    for (int nt = 0; nt < NTT; ++nt) {
      const v16bf bb = w2[(nt * KCL + kc) * 32 + lane];
      acc[nt] = __builtin_amdgcn_wmma_f32_16x16x32_bf16(
          false, a, false, bb, (short)0, acc[nt], false, false);
    }
  }

  // ---- epilogue: score[t] = v_b + sum_a tanh(pm+pq+pl)[t][a] * v_w[a] ----
  // C/D layout: lane(h,n), VGPR v  <->  (t = t0 + v + 8h, a = 16*nt + n).
  float part[8] = {0.f,0.f,0.f,0.f,0.f,0.f,0.f,0.f};
  #pragma unroll
  for (int nt = 0; nt < NTT; ++nt) {
    const int a_idx = nt * 16 + n;
    const float pqv = pq[b * AD + a_idx];
    const float vw  = v_w[a_idx];
    #pragma unroll
    for (int v = 0; v < 8; ++v)
      part[v] += tanhf(acc[nt][v] + pqv) * vw;
  }
  const float vb = v_b[0];
  #pragma unroll
  for (int v = 0; v < 8; ++v) {
    float r = part[v];                 // reduce over n within each lane half
    r += __shfl_xor(r, 1, 32);
    r += __shfl_xor(r, 2, 32);
    r += __shfl_xor(r, 4, 32);
    r += __shfl_xor(r, 8, 32);
    if (n == 0) scores[b * T_ + t0 + v + 8 * h] = r + vb;
  }
}

// ---------------- masked softmax over T, in place --------------------------
__global__ __launch_bounds__(256)
void lsa_softmax_kernel(const unsigned char* __restrict__ mask, float* __restrict__ sa) {
  const int b = blockIdx.x, tid = threadIdx.x;
  __shared__ float red[256];
  float* srow = sa + (size_t)b * T_;
  const unsigned char* mrow = mask + (size_t)b * T_;
  float mx = -INFINITY;
  for (int t = tid; t < T_; t += 256) {
    const float s = mrow[t] ? srow[t] : -INFINITY;
    mx = fmaxf(mx, s);
  }
  red[tid] = mx; __syncthreads();
  for (int o = 128; o; o >>= 1) { if (tid < o) red[tid] = fmaxf(red[tid], red[tid + o]); __syncthreads(); }
  mx = red[0]; __syncthreads();
  float sum = 0.f;
  for (int t = tid; t < T_; t += 256) {
    const float s = mrow[t] ? srow[t] : -INFINITY;
    sum += expf(s - mx);
  }
  red[tid] = sum; __syncthreads();
  for (int o = 128; o; o >>= 1) { if (tid < o) red[tid] += red[tid + o]; __syncthreads(); }
  const float inv = 1.0f / red[0]; __syncthreads();
  for (int t = tid; t < T_; t += 256) {
    const float s = mrow[t] ? srow[t] : -INFINITY;
    srow[t] = expf(s - mx) * inv;
  }
}

// ---------------- context: deterministic two-stage reduction ---------------
__global__ __launch_bounds__(256)
void lsa_ctx_partial_kernel(const float* __restrict__ memory, const float* __restrict__ align,
                            float* __restrict__ partial) {
  const int d   = blockIdx.x * 256 + threadIdx.x;  // 0..511
  const int tch = blockIdx.y;
  const int b   = blockIdx.z;
  const float* arow = align + (size_t)b * T_ + tch * TPC;
  const float* mcol = memory + ((size_t)b * T_ + tch * TPC) * MD + d;
  float acc = 0.f;
  for (int t = 0; t < TPC; ++t) acc = fmaf(arow[t], mcol[(size_t)t * MD], acc);
  partial[((size_t)tch * B_ + b) * MD + d] = acc;
}

__global__ __launch_bounds__(256)
void lsa_ctx_reduce_kernel(const float* __restrict__ partial, float* __restrict__ ctx) {
  const int i = blockIdx.x * 256 + threadIdx.x;    // i = b*MD + d
  float acc = 0.f;
  #pragma unroll
  for (int tch = 0; tch < TCH; ++tch) acc += partial[(size_t)tch * B_ * MD + i];
  ctx[i] = acc;
}

// ---------------------------------------------------------------------------
extern "C" void kernel_launch(void* const* d_in, const int* in_sizes, int n_in,
                              void* d_out, int out_size, void* d_ws, size_t ws_size,
                              hipStream_t stream) {
  (void)in_sizes; (void)n_in; (void)out_size; (void)ws_size;
  const float*         query    = (const float*)d_in[0];
  const float*         memory   = (const float*)d_in[1];
  const float*         aligncat = (const float*)d_in[2];
  const unsigned char* mask     = (const unsigned char*)d_in[3];  // numpy bool_ = 1 byte
  const float*         Wq       = (const float*)d_in[4];
  const float*         bq       = (const float*)d_in[5];
  const float*         Wm       = (const float*)d_in[6];
  const float*         bm       = (const float*)d_in[7];
  const float*         conv_w   = (const float*)d_in[8];
  const float*         Wl       = (const float*)d_in[9];
  const float*         v_w      = (const float*)d_in[10];
  const float*         v_b      = (const float*)d_in[11];

  float* ctx   = (float*)d_out;              // [B][MD]
  float* align = (float*)d_out + B_ * MD;    // [B][T]: scores -> softmax in place

  char* ws = (char*)d_ws;
  float*          pq      = (float*)ws;                                   // 32 KB
  unsigned short* wmfrag  = (unsigned short*)(ws + 32768);                // 128 KB bf16 frags
  unsigned short* w2frag  = (unsigned short*)(ws + 32768 + 131072);       // 16 KB bf16 frags
  float*          partial = (float*)(ws + 32768 + 131072 + 16384);        // 1 MB

  lsa_pq_kernel     <<<B_, AD, 0, stream>>>(query, Wq, bq, bm, pq);
  lsa_wmfrag_kernel <<<256, 256, 0, stream>>>(Wm, wmfrag);
  lsa_w2frag_kernel <<<32, 256, 0, stream>>>(conv_w, Wl, w2frag);
  lsa_score_kernel  <<<TOTAL_TILES / WPB, 32 * WPB, 0, stream>>>(
      memory, aligncat, pq, wmfrag, w2frag, v_w, v_b, align);
  lsa_softmax_kernel<<<B_, 256, 0, stream>>>(mask, align);
  lsa_ctx_partial_kernel<<<dim3(MD / 256, TCH, B_), 256, 0, stream>>>(memory, align, partial);
  lsa_ctx_reduce_kernel <<<B_ * MD / 256, 256, 0, stream>>>(partial, ctx);
}
